// PoseMixtureVAE_858993459257
// MI455X (gfx1250) — compile-verified, matrix operand order checked by reading, not run on hardware
//
#include <hip/hip_runtime.h>
#include <hip/hip_bf16.h>

// ---------------------------------------------------------------------------
// PoseMixtureVAE forward for MI455X (gfx1250, wave32, WMMA).
// All GEMMs use v_wmma_f32_16x16x32_bf16 (bf16 in, f32 accumulate).
// A and B operands are stored FRAGMENT-MAJOR in scratch: each 16x32 bf16
// fragment occupies 1024 contiguous bytes, 32 bytes per lane, so a fragment
// load is two global_load_b128 from (base + lane*32) -- no per-load address
// arithmetic, one A fragment's 2 b128 loads feed up to 6 WMMAs.
// ---------------------------------------------------------------------------

#define BATCH   4096
#define FRAME   267
#define LATENT  32
#define HIDDEN  256
#define GATE_H  64
#define EXPERTS 6
#define OUTD    267       // decoder output width

#define KP_ENC  544       // pad 534 / 523 -> 17*32
#define KP_D0   320       // pad 299 -> 10*32
#define KP_D1   288       // 9*32 exact
#define NC_OUT2 272       // pad 267 -> 17*16 column tiles

#define NK_ENC  (KP_ENC / 32)   // 17
#define NK_D0   (KP_D0 / 32)    // 10
#define NK_D1   (KP_D1 / 32)    // 9
#define MTILES  (BATCH / 16)    // 256

typedef __attribute__((ext_vector_type(16))) __bf16    v16bf;
typedef __attribute__((ext_vector_type(8)))  float     v8f;
typedef __attribute__((ext_vector_type(4)))  unsigned  v4u;

__device__ __forceinline__ float eluf(float x) {
  return x > 0.f ? x : (__expf(x) - 1.f);
}

__device__ __forceinline__ v8f zero8() {
  v8f z;
#pragma unroll
  for (int i = 0; i < 8; ++i) z[i] = 0.f;
  return z;
}

// Fragment-major element index. For matrix element (m, k) of a buffer with
// nK K-steps: tile = (m>>4)*nK + (k>>5); within the 16x32 tile the element
// lives at lane = (half<<4)|(m&15), dword v, per the ISA 16-bit A/B layout:
//   pair p = (k&31)>>1 ;  v = ((p>>3)<<2)|(p&3) ;  half = (p>>2)&1
// Tile is 512 bf16 = 1024 bytes; lane slice is 16 bf16 = 32 bytes.
__device__ __forceinline__ size_t frag_idx(int m, int k, int nK) {
  int mTile = m >> 4, lm = m & 15;
  int ks = k >> 5, kk = k & 31;
  int p = kk >> 1, lo = kk & 1;
  int half = (p >> 2) & 1;
  int v = ((p >> 3) << 2) | (p & 3);
  int lane = (half << 4) | lm;
  return ((size_t)(mTile * nK + ks) * 32 + lane) * 16 + (size_t)v * 2 + lo;
}

// Load one 16x32 bf16 fragment: two b128 loads per lane, contiguous.
__device__ __forceinline__ v16bf load_frag(const v4u* buf, size_t tileIdx, int lane) {
  union { v4u q[2]; v16bf v; } f;
  const v4u* p = buf + (tileIdx * 32 + lane) * 2;
  f.q[0] = p[0];
  f.q[1] = p[1];
  return f.v;
}

#define WMMA_BF16(a, b, c) \
  __builtin_amdgcn_wmma_f32_16x16x32_bf16(false, (a), false, (b), (short)0, (c), false, false)

// ---------------------------------------------------------------------------
// Packing kernels (elementwise, bandwidth-trivial; swizzle into frag layout).
// ---------------------------------------------------------------------------

// Zero-fill a padded activation buffer and copy up to two f32 sources in.
__global__ void k_pack_act(__hip_bfloat16* dst, int Kp, int nK,
                           const float* s1, int C1, int off1,
                           const float* s2, int C2, int off2) {
  int tid = blockIdx.x * blockDim.x + threadIdx.x;
  if (tid >= BATCH * Kp) return;
  int row = tid / Kp, col = tid % Kp;
  float v = 0.f;
  if (s1 && col >= off1 && col < off1 + C1) v = s1[(size_t)row * C1 + (col - off1)];
  else if (s2 && col >= off2 && col < off2 + C2) v = s2[(size_t)row * C2 + (col - off2)];
  dst[frag_idx(row, col, nK)] = __float2bfloat16(v);
}

// f32 [E][K][N] weights -> bf16 B-fragment layout (N plays the "m" role).
__global__ void k_pack_w(const float* src, __hip_bfloat16* dst, int K, int N,
                         int Kp, int Nc, int nK) {
  int e = blockIdx.z;
  int tid = blockIdx.x * blockDim.x + threadIdx.x;
  if (tid >= Kp * Nc) return;
  int k = tid / Nc, n = tid % Nc;
  float v = (k < K && n < N) ? src[(size_t)e * K * N + (size_t)k * N + n] : 0.f;
  size_t expStride = (size_t)(Nc / 16) * nK * 512;   // bf16 elements per expert
  dst[expStride * e + frag_idx(n, k, nK)] = __float2bfloat16(v);
}

// ---------------------------------------------------------------------------
// Encoder hidden layers: 16x64 tile per wave (one A fragment feeds 4 WMMAs).
// grid = (MTILES, HIDDEN/64), block = 32.
// ---------------------------------------------------------------------------
__global__ __launch_bounds__(32) void k_gemm_enc(
    const v4u* A, const v4u* W, int nK, const float* bias,
    __hip_bfloat16* dst, int dstColOff, int dstNK) {
  int lane = threadIdx.x, lm = lane & 15, half = lane >> 4;
  int mTile = blockIdx.x, nTile0 = blockIdx.y * 4;
  v8f acc[4];
#pragma unroll
  for (int t = 0; t < 4; ++t) acc[t] = zero8();
  for (int ks = 0; ks < nK; ++ks) {
    v16bf a = load_frag(A, (size_t)mTile * nK + ks, lane);
#pragma unroll
    for (int t = 0; t < 4; ++t) {
      v16bf b = load_frag(W, (size_t)(nTile0 + t) * nK + ks, lane);
      acc[t] = WMMA_BF16(a, b, acc[t]);
    }
  }
#pragma unroll
  for (int t = 0; t < 4; ++t) {
    int n = (nTile0 + t) * 16 + lm;
#pragma unroll
    for (int r = 0; r < 8; ++r) {
      int row = mTile * 16 + half * 8 + r;
      float v = eluf(acc[t][r] + bias[n]);
      dst[frag_idx(row, dstColOff + n, dstNK)] = __float2bfloat16(v);
    }
  }
}

// ---------------------------------------------------------------------------
// Encoder head: mu / logvar (N=32 -> 2 tiles each, 4 accumulators), then
// z = mu + eps*exp(0.5*lv), fanned out to the three decoder concat buffers.
// grid = MTILES, block = 32.
// ---------------------------------------------------------------------------
__global__ __launch_bounds__(32) void k_enc_head(
    const v4u* A, const v4u* Wmu, const v4u* Wlv,
    const float* bmu, const float* blv, const float* eps,
    float* out_mu, float* out_lv, float* zf,
    __hip_bfloat16* A0, __hip_bfloat16* A1, __hip_bfloat16* A2) {
  int lane = threadIdx.x, lm = lane & 15, half = lane >> 4;
  int mTile = blockIdx.x;
  v8f acc[4];                     // mu tile0, mu tile1, lv tile0, lv tile1
#pragma unroll
  for (int t = 0; t < 4; ++t) acc[t] = zero8();
  for (int ks = 0; ks < NK_ENC; ++ks) {
    v16bf a = load_frag(A, (size_t)mTile * NK_ENC + ks, lane);
#pragma unroll
    for (int t = 0; t < 2; ++t) {
      v16bf bm = load_frag(Wmu, (size_t)t * NK_ENC + ks, lane);
      v16bf bl = load_frag(Wlv, (size_t)t * NK_ENC + ks, lane);
      acc[t]     = WMMA_BF16(a, bm, acc[t]);
      acc[2 + t] = WMMA_BF16(a, bl, acc[2 + t]);
    }
  }
#pragma unroll
  for (int t = 0; t < 2; ++t) {
    int n = t * 16 + lm;
#pragma unroll
    for (int r = 0; r < 8; ++r) {
      int row = mTile * 16 + half * 8 + r;
      float mu = acc[t][r] + bmu[n];
      float lv = acc[2 + t][r] + blv[n];
      float z  = mu + eps[(size_t)row * LATENT + n] * __expf(0.5f * lv);
      out_mu[(size_t)row * LATENT + n] = mu;
      out_lv[(size_t)row * LATENT + n] = lv;
      zf[(size_t)row * LATENT + n]     = z;
      __hip_bfloat16 zb = __float2bfloat16(z);
      A0[frag_idx(row, n, NK_D0)] = zb;
      A1[frag_idx(row, n, NK_D1)] = zb;
      A2[frag_idx(row, n, NK_D1)] = zb;
    }
  }
}

// ---------------------------------------------------------------------------
// Gate MLP + softmax, one thread per sample (<1% of FLOPs; weight addresses
// are wave-uniform -> scalar loads).
// ---------------------------------------------------------------------------
__global__ __launch_bounds__(128) void k_gate(
    const float* zf, const float* c,
    const float* w0, const float* b0, const float* w1, const float* b1,
    const float* w2, const float* b2, float* coeff) {
  int n = blockIdx.x * blockDim.x + threadIdx.x;
  if (n >= BATCH) return;
  float h0[GATE_H];
  for (int j = 0; j < GATE_H; ++j) h0[j] = b0[j];
  for (int i = 0; i < LATENT; ++i) {
    float v = zf[(size_t)n * LATENT + i];
    const float* wr = w0 + (size_t)i * GATE_H;
    for (int j = 0; j < GATE_H; ++j) h0[j] += v * wr[j];
  }
  for (int i = 0; i < FRAME; ++i) {
    float v = c[(size_t)n * FRAME + i];
    const float* wr = w0 + (size_t)(LATENT + i) * GATE_H;
    for (int j = 0; j < GATE_H; ++j) h0[j] += v * wr[j];
  }
  for (int j = 0; j < GATE_H; ++j) h0[j] = eluf(h0[j]);
  float h1[GATE_H];
  for (int j = 0; j < GATE_H; ++j) h1[j] = b1[j];
  for (int i = 0; i < GATE_H; ++i) {
    float v = h0[i];
    const float* wr = w1 + (size_t)i * GATE_H;
    for (int j = 0; j < GATE_H; ++j) h1[j] += v * wr[j];
  }
  float lg[EXPERTS];
  for (int e = 0; e < EXPERTS; ++e) lg[e] = b2[e];
  for (int i = 0; i < GATE_H; ++i) {
    float v = eluf(h1[i]);
    const float* wr = w2 + (size_t)i * EXPERTS;
    for (int e = 0; e < EXPERTS; ++e) lg[e] += v * wr[e];
  }
  float mx = lg[0];
  for (int e = 1; e < EXPERTS; ++e) mx = fmaxf(mx, lg[e]);
  float s = 0.f;
  for (int e = 0; e < EXPERTS; ++e) { lg[e] = __expf(lg[e] - mx); s += lg[e]; }
  float inv = 1.f / s;
  for (int e = 0; e < EXPERTS; ++e) coeff[(size_t)n * EXPERTS + e] = lg[e] * inv;
}

// ---------------------------------------------------------------------------
// Soft-MoE decoder layer: out[n,o] = act( sum_e coeff[n,e]*((A@W_e)[n,o]+b_e[o]) )
// One wave per 16x16 tile; one A fragment feeds 6 expert WMMAs (6 live f32
// accumulator fragments). Coefficient blend after the K-reduction.
// grid = (MTILES, Nc/16), block = 32.
// ---------------------------------------------------------------------------
__global__ __launch_bounds__(32) void k_moe(
    const v4u* A, int nK, const v4u* W, size_t wTileStride,
    const float* bias, int Nout, const float* coeff, int doElu,
    __hip_bfloat16* dstB, int dstColOff, int dstNK, float* dstF) {
  int lane = threadIdx.x, lm = lane & 15, half = lane >> 4;
  int mTile = blockIdx.x, nTile = blockIdx.y;
  v8f acc[EXPERTS];
#pragma unroll
  for (int e = 0; e < EXPERTS; ++e) acc[e] = zero8();
  for (int ks = 0; ks < nK; ++ks) {
    v16bf a = load_frag(A, (size_t)mTile * nK + ks, lane);
#pragma unroll
    for (int e = 0; e < EXPERTS; ++e) {
      v16bf b = load_frag(W, wTileStride * e + (size_t)nTile * nK + ks, lane);
      acc[e] = WMMA_BF16(a, b, acc[e]);
    }
  }
  int n = nTile * 16 + lm;
#pragma unroll
  for (int r = 0; r < 8; ++r) {
    int row = mTile * 16 + half * 8 + r;
    float s = 0.f;
#pragma unroll
    for (int e = 0; e < EXPERTS; ++e) {
      float bv = (n < Nout) ? bias[(size_t)e * Nout + n] : 0.f;
      s += coeff[(size_t)row * EXPERTS + e] * (acc[e][r] + bv);
    }
    if (doElu) s = eluf(s);
    if (dstB) dstB[frag_idx(row, dstColOff + n, dstNK)] = __float2bfloat16(s);
    if (dstF && n < Nout) dstF[(size_t)row * Nout + n] = s;
  }
}

// ---------------------------------------------------------------------------
// Host launcher
// ---------------------------------------------------------------------------
extern "C" void kernel_launch(void* const* d_in, const int* in_sizes, int n_in,
                              void* d_out, int out_size, void* d_ws, size_t ws_size,
                              hipStream_t stream) {
  (void)in_sizes; (void)n_in; (void)out_size; (void)ws_size;

  const float* x       = (const float*)d_in[0];
  const float* c       = (const float*)d_in[1];
  const float* eps     = (const float*)d_in[2];
  const float* enc_w1  = (const float*)d_in[3];
  const float* enc_b1  = (const float*)d_in[4];
  const float* enc_w2  = (const float*)d_in[5];
  const float* enc_b2  = (const float*)d_in[6];
  const float* enc_wmu = (const float*)d_in[7];
  const float* enc_bmu = (const float*)d_in[8];
  const float* enc_wlv = (const float*)d_in[9];
  const float* enc_blv = (const float*)d_in[10];
  const float* g_w0    = (const float*)d_in[11];
  const float* g_b0    = (const float*)d_in[12];
  const float* g_w1    = (const float*)d_in[13];
  const float* g_b1    = (const float*)d_in[14];
  const float* g_w2    = (const float*)d_in[15];
  const float* g_b2    = (const float*)d_in[16];
  const float* w0      = (const float*)d_in[17];
  const float* b0      = (const float*)d_in[18];
  const float* w1      = (const float*)d_in[19];
  const float* b1      = (const float*)d_in[20];
  const float* w2      = (const float*)d_in[21];
  const float* b2      = (const float*)d_in[22];

  float* out       = (float*)d_out;
  float* out_layer = out;
  float* out_mu    = out + (size_t)BATCH * OUTD;
  float* out_lv    = out_mu + (size_t)BATCH * LATENT;

  // workspace bump allocator (256B aligned); total ~24 MB
  size_t off = 0;
  auto alloc = [&](size_t bytes) -> void* {
    off = (off + 255) & ~(size_t)255;
    void* p = (char*)d_ws + off;
    off += bytes;
    return p;
  };
  // A buffers: MTILES * nK tiles * 1024 B. W buffers: nTiles * nK * 1024 B / expert.
  __hip_bfloat16* Aenc1 = (__hip_bfloat16*)alloc((size_t)MTILES * NK_ENC * 1024);
  __hip_bfloat16* Aenc2 = (__hip_bfloat16*)alloc((size_t)MTILES * NK_ENC * 1024);
  __hip_bfloat16* Aenc3 = (__hip_bfloat16*)alloc((size_t)MTILES * NK_ENC * 1024);
  __hip_bfloat16* A0    = (__hip_bfloat16*)alloc((size_t)MTILES * NK_D0 * 1024);
  __hip_bfloat16* A1    = (__hip_bfloat16*)alloc((size_t)MTILES * NK_D1 * 1024);
  __hip_bfloat16* A2    = (__hip_bfloat16*)alloc((size_t)MTILES * NK_D1 * 1024);
  __hip_bfloat16* Wenc1 = (__hip_bfloat16*)alloc((size_t)(HIDDEN / 16) * NK_ENC * 1024);
  __hip_bfloat16* Wenc2 = (__hip_bfloat16*)alloc((size_t)(HIDDEN / 16) * NK_ENC * 1024);
  __hip_bfloat16* Wmu   = (__hip_bfloat16*)alloc((size_t)(LATENT / 16) * NK_ENC * 1024);
  __hip_bfloat16* Wlv   = (__hip_bfloat16*)alloc((size_t)(LATENT / 16) * NK_ENC * 1024);
  __hip_bfloat16* Wd0   = (__hip_bfloat16*)alloc((size_t)EXPERTS * (HIDDEN / 16) * NK_D0 * 1024);
  __hip_bfloat16* Wd1   = (__hip_bfloat16*)alloc((size_t)EXPERTS * (HIDDEN / 16) * NK_D1 * 1024);
  __hip_bfloat16* Wd2   = (__hip_bfloat16*)alloc((size_t)EXPERTS * (NC_OUT2 / 16) * NK_D1 * 1024);
  float*          zf    = (float*)alloc((size_t)BATCH * LATENT * 4);
  float*          coeff = (float*)alloc((size_t)BATCH * EXPERTS * 4);

  const int TB = 256;
  auto cdiv = [](int a, int b) { return (a + b - 1) / b; };

  // --- pack activations (x, c into padded bf16 frag-major concat buffers) ---
  k_pack_act<<<cdiv(BATCH * KP_ENC, TB), TB, 0, stream>>>(Aenc1, KP_ENC, NK_ENC, x, FRAME, 0, c, FRAME, FRAME);
  k_pack_act<<<cdiv(BATCH * KP_ENC, TB), TB, 0, stream>>>(Aenc2, KP_ENC, NK_ENC, x, FRAME, 0, nullptr, 0, 0);
  k_pack_act<<<cdiv(BATCH * KP_ENC, TB), TB, 0, stream>>>(Aenc3, KP_ENC, NK_ENC, x, FRAME, 0, nullptr, 0, 0);
  k_pack_act<<<cdiv(BATCH * KP_D0,  TB), TB, 0, stream>>>(A0, KP_D0, NK_D0, c, FRAME, LATENT, nullptr, 0, 0);

  // --- pack weights into bf16 B-fragment layout ---
  k_pack_w<<<dim3(cdiv(KP_ENC * HIDDEN, TB), 1, 1), TB, 0, stream>>>(enc_w1, Wenc1, 2 * FRAME, HIDDEN, KP_ENC, HIDDEN, NK_ENC);
  k_pack_w<<<dim3(cdiv(KP_ENC * HIDDEN, TB), 1, 1), TB, 0, stream>>>(enc_w2, Wenc2, FRAME + HIDDEN, HIDDEN, KP_ENC, HIDDEN, NK_ENC);
  k_pack_w<<<dim3(cdiv(KP_ENC * LATENT, TB), 1, 1), TB, 0, stream>>>(enc_wmu, Wmu, FRAME + HIDDEN, LATENT, KP_ENC, LATENT, NK_ENC);
  k_pack_w<<<dim3(cdiv(KP_ENC * LATENT, TB), 1, 1), TB, 0, stream>>>(enc_wlv, Wlv, FRAME + HIDDEN, LATENT, KP_ENC, LATENT, NK_ENC);
  k_pack_w<<<dim3(cdiv(KP_D0 * HIDDEN, TB), 1, EXPERTS), TB, 0, stream>>>(w0, Wd0, LATENT + FRAME, HIDDEN, KP_D0, HIDDEN, NK_D0);
  k_pack_w<<<dim3(cdiv(KP_D1 * HIDDEN, TB), 1, EXPERTS), TB, 0, stream>>>(w1, Wd1, LATENT + HIDDEN, HIDDEN, KP_D1, HIDDEN, NK_D1);
  k_pack_w<<<dim3(cdiv(KP_D1 * NC_OUT2, TB), 1, EXPERTS), TB, 0, stream>>>(w2, Wd2, LATENT + HIDDEN, OUTD, KP_D1, NC_OUT2, NK_D1);

  // --- encoder hidden layers (WMMA, fused ELU, bf16 into next concat buf) ---
  k_gemm_enc<<<dim3(MTILES, HIDDEN / 64), 32, 0, stream>>>(
      (const v4u*)Aenc1, (const v4u*)Wenc1, NK_ENC, enc_b1, Aenc2, FRAME, NK_ENC);
  k_gemm_enc<<<dim3(MTILES, HIDDEN / 64), 32, 0, stream>>>(
      (const v4u*)Aenc2, (const v4u*)Wenc2, NK_ENC, enc_b2, Aenc3, FRAME, NK_ENC);

  // --- encoder head: mu / logvar / z (z fanned into decoder concat bufs) ---
  k_enc_head<<<MTILES, 32, 0, stream>>>(
      (const v4u*)Aenc3, (const v4u*)Wmu, (const v4u*)Wlv,
      enc_bmu, enc_blv, eps, out_mu, out_lv, zf, A0, A1, A2);

  // --- gate MLP + softmax ---
  k_gate<<<cdiv(BATCH, 128), 128, 0, stream>>>(zf, c, g_w0, g_b0, g_w1, g_b1, g_w2, g_b2, coeff);

  // --- soft-MoE decoder layers ---
  k_moe<<<dim3(MTILES, HIDDEN / 16), 32, 0, stream>>>(
      (const v4u*)A0, NK_D0, (const v4u*)Wd0, (size_t)(HIDDEN / 16) * NK_D0,
      b0, HIDDEN, coeff, 1, A1, LATENT, NK_D1, nullptr);
  k_moe<<<dim3(MTILES, HIDDEN / 16), 32, 0, stream>>>(
      (const v4u*)A1, NK_D1, (const v4u*)Wd1, (size_t)(HIDDEN / 16) * NK_D1,
      b1, HIDDEN, coeff, 1, A2, LATENT, NK_D1, nullptr);
  k_moe<<<dim3(MTILES, NC_OUT2 / 16), 32, 0, stream>>>(
      (const v4u*)A2, NK_D1, (const v4u*)Wd2, (size_t)(NC_OUT2 / 16) * NK_D1,
      b2, OUTD, coeff, 0, nullptr, 0, 0, out_layer);
}
